// MoELayer_3487513444667
// MI455X (gfx1250) — compile-verified, matrix operand order checked by reading, not run on hardware
//
#include <hip/hip_runtime.h>
#include <hip/hip_bf16.h>
#include <math.h>

typedef unsigned short u16;
typedef unsigned int u32;

#define Hdim 1024
#define Idim 2048
#define Edim 8
#define Tdim 4096
#define ROUTER_BLOCKS (Tdim / 256)

typedef __attribute__((ext_vector_type(16))) __bf16 v16bf;
typedef __attribute__((ext_vector_type(8)))  float  v8f;

union FragU { uint4 q[2]; v16bf v; };

__device__ __forceinline__ u16 f2bf(float x) {
    u32 u = __float_as_uint(x);
    u32 r = u + 0x7FFFu + ((u >> 16) & 1u);   // round-to-nearest-even
    return (u16)(r >> 16);
}

// silu(g) = g * sigmoid(g); fast path: v_exp_f32 + v_rcp_f32 (no IEEE div expansion)
__device__ __forceinline__ float silu_fast(float g) {
    return g * __builtin_amdgcn_rcpf(1.f + __expf(-g));
}

// Load a 16-element bf16 WMMA fragment from global memory.
// pkb = row pointer already offset by per-lane kb (0 or 8); k0 in elements.
__device__ __forceinline__ v16bf ldfragG(const u16* __restrict__ pkb, int k0) {
    FragU f;
    f.q[0] = *(const uint4*)(pkb + k0);
    f.q[1] = *(const uint4*)(pkb + k0 + 16);
    return f.v;
}

// ---------------------------------------------------------------------------
// 1) x fp32 -> bf16 (row major, [T][H])
// ---------------------------------------------------------------------------
__global__ void moe_cvt_x(const float* __restrict__ src, u16* __restrict__ dst) {
    size_t i = ((size_t)blockIdx.x * 256 + threadIdx.x) * 4;
    float4 v = *(const float4*)(src + i);
    uint2 o;
    o.x = ((u32)f2bf(v.y) << 16) | (u32)f2bf(v.x);
    o.y = ((u32)f2bf(v.w) << 16) | (u32)f2bf(v.z);
    *(uint2*)(dst + i) = o;
}

// ---------------------------------------------------------------------------
// 2) weight transpose + convert: src [E][R][C] f32 -> dst [E][C][R] bf16
// ---------------------------------------------------------------------------
__global__ void moe_transpose_cvt(const float* __restrict__ src, u16* __restrict__ dst,
                                  int R, int C) {
    __shared__ float tile[32][33];
    const int e = blockIdx.z;
    const float* s = src + (size_t)e * R * C;
    u16* d = dst + (size_t)e * R * C;
    const int c0 = blockIdx.x * 32, r0 = blockIdx.y * 32;
    const int tx = threadIdx.x, ty = threadIdx.y;   // 32 x 8
    #pragma unroll
    for (int j = 0; j < 32; j += 8)
        tile[ty + j][tx] = s[(size_t)(r0 + ty + j) * C + c0 + tx];
    __syncthreads();
    #pragma unroll
    for (int j = 0; j < 32; j += 8)
        d[(size_t)(c0 + ty + j) * R + r0 + tx] = f2bf(tile[tx][ty + j]);
}

// ---------------------------------------------------------------------------
// 3) Router: logits, softmax, top-2, normalized weights, P partial sums
// ---------------------------------------------------------------------------
__global__ void __launch_bounds__(256) moe_router(
    const float* __restrict__ x, const float* __restrict__ rw,
    int* __restrict__ topI, float* __restrict__ topW, float* __restrict__ Ppart) {
    __shared__ float rw_s[Hdim * Edim];     // 32 KB
    __shared__ float wred[8][8];
    const int tid = threadIdx.x;

    for (int i = tid * 4; i < Hdim * Edim; i += 256 * 4)
        *(float4*)(rw_s + i) = *(const float4*)(rw + i);
    __syncthreads();

    const int t = blockIdx.x * 256 + tid;
    const float* xr = x + (size_t)t * Hdim;
    float acc[8] = {0.f, 0.f, 0.f, 0.f, 0.f, 0.f, 0.f, 0.f};
    for (int h = 0; h < Hdim; h += 4) {
        float4 xv = *(const float4*)(xr + h);
        #pragma unroll
        for (int c = 0; c < 4; ++c) {
            float xs = (&xv.x)[c];
            #pragma unroll
            for (int e = 0; e < 8; ++e)
                acc[e] = fmaf(xs, rw_s[(h + c) * 8 + e], acc[e]);
        }
    }
    // softmax over 8
    float m = acc[0];
    #pragma unroll
    for (int e = 1; e < 8; ++e) m = fmaxf(m, acc[e]);
    float p[8], s = 0.f;
    #pragma unroll
    for (int e = 0; e < 8; ++e) { p[e] = __expf(acc[e] - m); s += p[e]; }
    float inv = 1.f / s;
    #pragma unroll
    for (int e = 0; e < 8; ++e) p[e] *= inv;
    // top-2 (first index wins ties, matching lax.top_k)
    int i0 = 0; float b0 = p[0];
    #pragma unroll
    for (int e = 1; e < 8; ++e) if (p[e] > b0) { b0 = p[e]; i0 = e; }
    int i1 = (i0 == 0) ? 1 : 0; float b1 = p[i1];
    #pragma unroll
    for (int e = 0; e < 8; ++e)
        if (e != i0 && p[e] > b1) { b1 = p[e]; i1 = e; }
    float ws = 1.f / (b0 + b1);
    topI[t * 2 + 0] = i0; topI[t * 2 + 1] = i1;
    topW[t * 2 + 0] = b0 * ws; topW[t * 2 + 1] = b1 * ws;

    // block-level partial sum of probs (for aux loss P)
    const int lane = tid & 31, wave = tid >> 5;
    #pragma unroll
    for (int e = 0; e < 8; ++e) {
        float v = p[e];
        for (int o = 16; o > 0; o >>= 1) v += __shfl_xor(v, o, 32);
        if (lane == 0) wred[wave][e] = v;
    }
    __syncthreads();
    if (tid < 8) {
        float v = 0.f;
        #pragma unroll
        for (int w = 0; w < 8; ++w) v += wred[w][tid];
        Ppart[blockIdx.x * 8 + tid] = v;
    }
}

// ---------------------------------------------------------------------------
// 4) Deterministic compaction: wave e builds token list for expert e
// ---------------------------------------------------------------------------
__global__ void moe_compact(const int* __restrict__ topI,
                            int* __restrict__ lists, int* __restrict__ counts) {
    const int e = threadIdx.x >> 5;
    const int lane = threadIdx.x & 31;
    const u32 lmask = (1u << lane) - 1u;
    int offset = 0;
    for (int t0 = 0; t0 < Tdim; t0 += 32) {
        const int t = t0 + lane;
        const int a = topI[t * 2], b = topI[t * 2 + 1];
        const bool pred = (a == e) || (b == e);
        const int slot = (a == e) ? 0 : 1;
        const u32 mask = __builtin_amdgcn_ballot_w32(pred);
        if (pred) lists[e * Tdim + offset + __popc(mask & lmask)] = (t << 1) | slot;
        offset += __popc(mask);
    }
    if (lane == 0) counts[e] = offset;
}

// ---------------------------------------------------------------------------
// 5) Fused expert FFN: per (expert, 32-token tile); bf16 WMMA SwiGLU + down-proj
//    M=32 doubles WMMA:VMEM ratio vs M=16 (B fragments reused by 2 A tiles).
// ---------------------------------------------------------------------------
__global__ void __launch_bounds__(256) moe_ffn(
    const u16* __restrict__ xbf, const u16* __restrict__ gateT,
    const u16* __restrict__ upT, const u16* __restrict__ downT,
    const float* __restrict__ topW, const int* __restrict__ lists,
    const int* __restrict__ counts, float* __restrict__ ybuf) {
    __shared__ u16 h_lds[32 * 1024];    // 64 KB: half of h tile (32 tokens), bf16

    const int e = blockIdx.y;
    const int m0 = blockIdx.x * 32;
    const int count = counts[e];
    if (m0 >= count) return;
    int rows = count - m0; if (rows > 32) rows = 32;

    const int tid = threadIdx.x;
    const int wave = tid >> 5;
    const int lane = tid & 31;
    const int laneM = lane & 15;
    const int laneHi = lane >> 4;
    const int kb = laneHi << 3;         // 0 or 8 per WMMA bf16 fragment layout

    // Per-lane gather rows for the two A tiles (rows laneM and laneM+16).
    const int entA0 = (laneM < rows)      ? lists[e * Tdim + m0 + laneM]      : 0;
    const int entA1 = (laneM + 16 < rows) ? lists[e * Tdim + m0 + 16 + laneM] : 0;
    const u16* aRow0 = xbf + (size_t)(entA0 >> 1) * Hdim + kb;
    const u16* aRow1 = xbf + (size_t)(entA1 >> 1) * Hdim + kb;

    v8f zero = {};
    v8f yacc0[8], yacc1[8];
    #pragma unroll
    for (int nn = 0; nn < 8; ++nn) { yacc0[nn] = zero; yacc1[nn] = zero; }

    #pragma unroll
    for (int half = 0; half < 2; ++half) {
        const int iBase = half * 1024;
        // ---- Phase 1: g,u = x @ Wg, x @ Wu ; h = silu(g)*u -> LDS ----
        #pragma unroll 1
        for (int nn = 0; nn < 8; ++nn) {
            const int nt = wave + nn * 8;                 // n-tile within half
            const int nI = iBase + nt * 16 + laneM;       // column in I
            const u16* gRow = gateT + ((size_t)e * Idim + nI) * Hdim + kb;
            const u16* uRow = upT   + ((size_t)e * Idim + nI) * Hdim + kb;
            if (nn < 7) {   // prefetch next iteration's weight rows (global_prefetch_b8)
                __builtin_prefetch((const void*)(gRow + (size_t)128 * Hdim), 0, 0);
                __builtin_prefetch((const void*)(uRow + (size_t)128 * Hdim), 0, 0);
            }
            v8f g0 = zero, g1 = zero, u0 = zero, u1 = zero;
            #pragma unroll 2
            for (int k0 = 0; k0 < Hdim; k0 += 32) {
                v16bf a0 = ldfragG(aRow0, k0);
                v16bf a1 = ldfragG(aRow1, k0);
                v16bf bg = ldfragG(gRow, k0);
                v16bf bu = ldfragG(uRow, k0);
                g0 = __builtin_amdgcn_wmma_f32_16x16x32_bf16(false, a0, false, bg, (short)0, g0, false, false);
                g1 = __builtin_amdgcn_wmma_f32_16x16x32_bf16(false, a1, false, bg, (short)0, g1, false, false);
                u0 = __builtin_amdgcn_wmma_f32_16x16x32_bf16(false, a0, false, bu, (short)0, u0, false, false);
                u1 = __builtin_amdgcn_wmma_f32_16x16x32_bf16(false, a1, false, bu, (short)0, u1, false, false);
            }
            const int mB = laneHi * 8;
            #pragma unroll
            for (int r = 0; r < 8; ++r) {
                const float hv0 = silu_fast(g0[r]) * u0[r];
                const float hv1 = silu_fast(g1[r]) * u1[r];
                h_lds[(mB + r) * 1024 + nt * 16 + laneM] = f2bf(hv0);
                h_lds[(mB + r + 16) * 1024 + nt * 16 + laneM] = f2bf(hv1);
            }
        }
        __syncthreads();
        // ---- Phase 2 (partial over this I-half): y += h @ Wd ----
        #pragma unroll
        for (int nn = 0; nn < 8; ++nn) {
            const int nt = wave + nn * 8;                 // n-tile over H
            const int nH = nt * 16 + laneM;
            const u16* dRow = downT + ((size_t)e * Hdim + nH) * Idim + iBase + kb;
            const int aOff0 = laneM * 1024 + kb;
            const int aOff1 = (laneM + 16) * 1024 + kb;
            v8f acc0 = yacc0[nn];
            v8f acc1 = yacc1[nn];
            #pragma unroll 2
            for (int k0 = 0; k0 < 1024; k0 += 32) {
                FragU fa0, fa1;
                fa0.q[0] = *(const uint4*)&h_lds[aOff0 + k0];
                fa0.q[1] = *(const uint4*)&h_lds[aOff0 + k0 + 16];
                fa1.q[0] = *(const uint4*)&h_lds[aOff1 + k0];
                fa1.q[1] = *(const uint4*)&h_lds[aOff1 + k0 + 16];
                v16bf b = ldfragG(dRow, k0);
                acc0 = __builtin_amdgcn_wmma_f32_16x16x32_bf16(false, fa0.v, false, b, (short)0, acc0, false, false);
                acc1 = __builtin_amdgcn_wmma_f32_16x16x32_bf16(false, fa1.v, false, b, (short)0, acc1, false, false);
            }
            yacc0[nn] = acc0;
            yacc1[nn] = acc1;
        }
        __syncthreads();
    }

    // ---- Per-lane routing metadata for the 16 rows this lane stores ----
    int tok0[8], slot0[8], tok1[8], slot1[8];
    float w0[8], w1[8];
    #pragma unroll
    for (int r = 0; r < 8; ++r) {
        const int mA = laneHi * 8 + r;        // row in tile 0
        const int mBr = mA + 16;              // row in tile 1
        const int eA = (mA < rows)  ? lists[e * Tdim + m0 + mA]  : 0;
        const int eB = (mBr < rows) ? lists[e * Tdim + m0 + mBr] : 0;
        tok0[r] = eA >> 1; slot0[r] = eA & 1;
        tok1[r] = eB >> 1; slot1[r] = eB & 1;
        w0[r] = (mA < rows)  ? topW[tok0[r] * 2 + slot0[r]] : 0.f;
        w1[r] = (mBr < rows) ? topW[tok1[r] * 2 + slot1[r]] : 0.f;
    }

    // ---- Store weighted rows into per-(token,slot) buffer (no atomics) ----
    #pragma unroll
    for (int nn = 0; nn < 8; ++nn) {
        const int n0 = (wave + nn * 8) * 16;
        #pragma unroll
        for (int r = 0; r < 8; ++r) {
            const int mA = laneHi * 8 + r;
            if (mA < rows) {
                const size_t o = ((size_t)tok0[r] * 2 + slot0[r]) * Hdim + n0 + laneM;
                ybuf[o] = yacc0[nn][r] * w0[r];
            }
            if (mA + 16 < rows) {
                const size_t o = ((size_t)tok1[r] * 2 + slot1[r]) * Hdim + n0 + laneM;
                ybuf[o] = yacc1[nn][r] * w1[r];
            }
        }
    }
}

// ---------------------------------------------------------------------------
// 6) Combine the two expert slots per token
// ---------------------------------------------------------------------------
__global__ void moe_combine(const float* __restrict__ ybuf, float* __restrict__ out) {
    const size_t i = (size_t)blockIdx.x * 256 + threadIdx.x;    // over T*H/4
    const size_t t = i / (Hdim / 4);
    const size_t off = i % (Hdim / 4);
    float4 a = *((const float4*)(ybuf + t * 2 * Hdim) + off);
    float4 b = *((const float4*)(ybuf + t * 2 * Hdim + Hdim) + off);
    float4 o; o.x = a.x + b.x; o.y = a.y + b.y; o.z = a.z + b.z; o.w = a.w + b.w;
    *((float4*)out + i) = o;
}

// ---------------------------------------------------------------------------
// 7) Aux load-balancing loss
// ---------------------------------------------------------------------------
__global__ void moe_aux(const float* __restrict__ Ppart, const int* __restrict__ cnts,
                        float* __restrict__ out) {
    __shared__ float red[8];
    const int tid = threadIdx.x;
    if (tid < 8) {
        float P = 0.f;
        for (int b = 0; b < ROUTER_BLOCKS; ++b) P += Ppart[b * 8 + tid];
        P /= (float)Tdim;
        const float f = (float)cnts[tid] / (float)Tdim;
        red[tid] = f * P;
    }
    __syncthreads();
    if (tid == 0) {
        float s = 0.f;
        #pragma unroll
        for (int e = 0; e < 8; ++e) s += red[e];
        out[(size_t)Tdim * Hdim] = s * (float)Edim * 0.01f;
    }
}

// ---------------------------------------------------------------------------
extern "C" void kernel_launch(void* const* d_in, const int* in_sizes, int n_in,
                              void* d_out, int out_size, void* d_ws, size_t ws_size,
                              hipStream_t stream) {
    (void)in_sizes; (void)n_in; (void)out_size; (void)ws_size;
    const float* x  = (const float*)d_in[0];
    const float* rw = (const float*)d_in[1];
    const float* wg = (const float*)d_in[2];
    const float* wu = (const float*)d_in[3];
    const float* wd = (const float*)d_in[4];
    float* out = (float*)d_out;

    char* ws = (char*)d_ws;
    size_t off = 0;
    auto take = [&](size_t bytes) -> char* {
        char* p = ws + off;
        off += (bytes + 255) & ~(size_t)255;
        return p;
    };
    u16*   xbf   = (u16*)  take((size_t)Tdim * Hdim * 2);
    u16*   gateT = (u16*)  take((size_t)Edim * Idim * Hdim * 2);
    u16*   upT   = (u16*)  take((size_t)Edim * Idim * Hdim * 2);
    u16*   downT = (u16*)  take((size_t)Edim * Hdim * Idim * 2);
    int*   topI  = (int*)  take((size_t)Tdim * 2 * sizeof(int));
    float* topW  = (float*)take((size_t)Tdim * 2 * sizeof(float));
    int*   lists = (int*)  take((size_t)Edim * Tdim * sizeof(int));
    int*   cnts  = (int*)  take((size_t)Edim * sizeof(int));
    float* Ppart = (float*)take((size_t)ROUTER_BLOCKS * Edim * sizeof(float));
    float* ybuf  = (float*)take((size_t)Tdim * 2 * Hdim * sizeof(float));

    moe_cvt_x<<<(Tdim * Hdim / 4) / 256, 256, 0, stream>>>(x, xbf);
    moe_transpose_cvt<<<dim3(Idim / 32, Hdim / 32, Edim), dim3(32, 8), 0, stream>>>(wg, gateT, Hdim, Idim);
    moe_transpose_cvt<<<dim3(Idim / 32, Hdim / 32, Edim), dim3(32, 8), 0, stream>>>(wu, upT, Hdim, Idim);
    moe_transpose_cvt<<<dim3(Hdim / 32, Idim / 32, Edim), dim3(32, 8), 0, stream>>>(wd, downT, Idim, Hdim);
    moe_router<<<ROUTER_BLOCKS, 256, 0, stream>>>(x, rw, topI, topW, Ppart);
    moe_compact<<<1, 256, 0, stream>>>(topI, lists, cnts);
    moe_ffn<<<dim3(Tdim / 32, Edim), 256, 0, stream>>>(xbf, gateT, upT, downT, topW, lists, cnts, ybuf);
    moe_combine<<<(Tdim * Hdim / 4) / 256, 256, 0, stream>>>(ybuf, out);
    moe_aux<<<1, 32, 0, stream>>>(Ppart, cnts, out);
}
